// NearLoss_69217692942359
// MI455X (gfx1250) — compile-verified, matrix operand order checked by reading, not run on hardware
//
#include <hip/hip_runtime.h>
#include <hip/hip_bf16.h>

#ifndef __has_builtin
#define __has_builtin(x) 0
#endif

#define NUM 32768
#define DIM 2048
#define NCLS 512
#define EPSV 1e-6f

#define TILE_COLS 16          // columns per WG tile
#define N_TILES   (DIM / TILE_COLS)   // 128
#define ROW_GROUPS 4
#define ROWS_PER_WG (NUM / ROW_GROUPS) // 8192
#define CHUNK 64              // rows staged per async chunk

typedef __attribute__((ext_vector_type(16))) _Float16 v16h;
typedef __attribute__((ext_vector_type(8)))  float    v8f;

// Match the builtin's parameter type exactly: gcc-style vector of 4 ints.
typedef int v4i_vs __attribute__((__vector_size__(16)));
typedef __attribute__((address_space(1))) v4i_vs* g_v4i_ptr;  // global
typedef __attribute__((address_space(3))) v4i_vs* l_v4i_ptr;  // LDS

// ---------------- CDNA5 async copy to LDS (guarded) ----------------
#if __has_builtin(__builtin_amdgcn_global_load_async_to_lds_b128)
#define HAVE_ASYNC_LDS 1
#else
#define HAVE_ASYNC_LDS 0
#endif

__device__ __forceinline__ void async_copy16(const float* gp, float* lp) {
#if HAVE_ASYNC_LDS
  __builtin_amdgcn_global_load_async_to_lds_b128(
      (g_v4i_ptr)(unsigned long long)gp,
      (l_v4i_ptr)(unsigned int)(unsigned long long)lp,
      0, 0);
#else
  *(float4*)lp = *(const float4*)gp;
#endif
}

__device__ __forceinline__ void wait_async_le1() {
#if HAVE_ASYNC_LDS
#if __has_builtin(__builtin_amdgcn_s_wait_asynccnt)
  __builtin_amdgcn_s_wait_asynccnt(1);
#else
  asm volatile("s_wait_asynccnt 1" ::: "memory");
#endif
#endif
}
__device__ __forceinline__ void wait_async_le0() {
#if HAVE_ASYNC_LDS
#if __has_builtin(__builtin_amdgcn_s_wait_asynccnt)
  __builtin_amdgcn_s_wait_asynccnt(0);
#else
  asm volatile("s_wait_asynccnt 0" ::: "memory");
#endif
#endif
}

// ---------------- kernels ----------------

__global__ void k_init(float* ws, size_t n) {
  size_t i = (size_t)blockIdx.x * blockDim.x + threadIdx.x;
  size_t stride = (size_t)gridDim.x * blockDim.x;
  for (; i < n; i += stride) ws[i] = 0.0f;
}

__global__ void k_counts(const int* __restrict__ labels, float* __restrict__ counts) {
  int i = blockIdx.x * blockDim.x + threadIdx.x;
  if (i < NUM) unsafeAtomicAdd(&counts[labels[i]], 1.0f);
}

// Main streaming pass: per-class partial sums (per row-group slice) + sum of squares.
__global__ __launch_bounds__(256)
void k_main(const float* __restrict__ x, const int* __restrict__ labels,
            float* __restrict__ sums_slices, float* __restrict__ accum) {
  __shared__ float acc[NCLS * TILE_COLS];       // 32 KB class accumulator
  __shared__ float stage[2][CHUNK * TILE_COLS]; // 2 x 4 KB staging
  __shared__ float red[256];

  const int tid  = threadIdx.x;
  const int tile = blockIdx.x;      // 0..127
  const int g    = blockIdx.y;      // 0..3 row group
  const int col0 = tile * TILE_COLS;
  const int row0 = g * ROWS_PER_WG;
  const int nchunks = ROWS_PER_WG / CHUNK;    // 128

  for (int i = tid; i < NCLS * TILE_COLS; i += 256) acc[i] = 0.0f;
  __syncthreads();

  // issue: thread t copies 16B; row = t/4, 4-float segment = t%4
  const int srow = tid >> 2;
  const int sseg = tid & 3;

  // prefetch chunk 0 into buffer 0
  {
    const float* gp = x + (size_t)(row0 + srow) * DIM + col0 + sseg * 4;
    async_copy16(gp, &stage[0][srow * TILE_COLS + sseg * 4]);
  }

  float sq = 0.0f;
  for (int ch = 0; ch < nchunks; ++ch) {
    const int buf = ch & 1;
    if (ch + 1 < nchunks) {
      const float* gp = x + (size_t)(row0 + (ch + 1) * CHUNK + srow) * DIM + col0 + sseg * 4;
      async_copy16(gp, &stage[buf ^ 1][srow * TILE_COLS + sseg * 4]);
      wait_async_le1();   // chunk ch complete, ch+1 in flight
    } else {
      wait_async_le0();
    }
    __syncthreads();      // cross-wave visibility of staged data

    const int rbase = row0 + ch * CHUNK;
#pragma unroll
    for (int p = 0; p < CHUNK / 16; ++p) {   // 4 passes, 16 rows x 16 cols each
      const int r = p * 16 + (tid >> 4);
      const int c = tid & 15;
      const float v = stage[buf][r * TILE_COLS + c];
      const int cls = labels[rbase + r];     // broadcast across 16 lanes
      unsafeAtomicAdd(&acc[cls * TILE_COLS + c], v);  // ds_add_f32
      sq += v * v;
    }
    __syncthreads();      // done reading buf before it is refilled
  }

  // flush class accumulator to this row-group's private slice (plain stores)
  float* slice = sums_slices + (size_t)g * ((size_t)NCLS * DIM);
  for (int i = tid; i < NCLS * TILE_COLS; i += 256) {
    const int cls = i >> 4;
    const int c   = i & 15;
    slice[(size_t)cls * DIM + col0 + c] = acc[i];
  }

  // reduce sum of squares
  red[tid] = sq;
  __syncthreads();
  for (int s = 128; s > 0; s >>= 1) {
    if (tid < s) red[tid] += red[tid + s];
    __syncthreads();
  }
  if (tid == 0) unsafeAtomicAdd(accum, red[0]);
}

// Finalize: subtract sum_c ||S_c||^2 / cnt_c using WMMA Gram-diagonal.
// For 16 classes, diag(S S^T) over K=2048, f16 inputs / f32 accumulate.
// A-fragment (M=class,K) and B-fragment of S^T (K,N=class) have identical
// lane layout, so wmma(a, a, c) accumulates the Gram block.
__global__ __launch_bounds__(256)
void k_finalize(const float* __restrict__ sums_slices, const float* __restrict__ counts,
                float* __restrict__ accum) {
  __shared__ float lbuf[8 * 256];   // 8 waves x (8 vgprs x 32 lanes)
  const int tid  = threadIdx.x;
  const int w    = tid >> 5;
  const int lane = tid & 31;
  const int cb   = blockIdx.x * 16;

  const int m   = (lane < 16) ? lane : lane - 16;
  const int off = (lane < 16) ? 0 : 8;
  const size_t rowbase = (size_t)(cb + m) * DIM;

  v8f c = {0.f, 0.f, 0.f, 0.f, 0.f, 0.f, 0.f, 0.f};
  for (int j = 0; j < 8; ++j) {           // 64 K-chunks of 32, 8 per wave
    const int k0 = (w + 8 * j) * 32;
    v16h a;
#pragma unroll
    for (int q = 0; q < 8; ++q) {
      float v0 = 0.f, v1 = 0.f;
#pragma unroll
      for (int s = 0; s < ROW_GROUPS; ++s) {
        const float* sl = sums_slices + (size_t)s * ((size_t)NCLS * DIM);
        v0 += sl[rowbase + k0 + off + q];
        v1 += sl[rowbase + k0 + 16 + off + q];
      }
      a[q]     = (_Float16)v0;   // elems 0..7  : K = k0+off .. +7
      a[8 + q] = (_Float16)v1;   // elems 8..15 : K = k0+16+off .. +7
    }
    c = __builtin_amdgcn_wmma_f32_16x16x32_f16(
        false, a, false, a, (short)0, c, false, false);
  }

#pragma unroll
  for (int r = 0; r < 8; ++r) lbuf[w * 256 + r * 32 + lane] = c[r];
  __syncthreads();

  if (tid < 16) {
    // C layout: (M=r, N=lane) for lanes 0-15; (M=8+r, N=lane-16) for 16-31.
    float diag = 0.f;
    for (int ww = 0; ww < 8; ++ww)
      diag += (tid < 8) ? lbuf[ww * 256 + tid * 32 + tid]
                        : lbuf[ww * 256 + (tid - 8) * 32 + 16 + tid];
    const float cnt = counts[cb + tid];
    if (cnt > 0.5f) unsafeAtomicAdd(accum, -(diag / cnt));
  }
}

__global__ void k_write(const float* __restrict__ accum, float* __restrict__ out) {
  if (threadIdx.x == 0 && blockIdx.x == 0)
    out[0] = (accum[0] + (float)NUM * (float)DIM * EPSV * EPSV) / (float)NUM;
}

// ---------------- launcher ----------------
extern "C" void kernel_launch(void* const* d_in, const int* in_sizes, int n_in,
                              void* d_out, int out_size, void* d_ws, size_t ws_size,
                              hipStream_t stream) {
  const float* x      = (const float*)d_in[0];
  const int*   labels = (const int*)d_in[1];
  float* ws     = (float*)d_ws;
  float* accum  = ws;           // ws[0]
  float* counts = ws + 8;       // ws[8 .. 519]
  float* sums   = ws + 1024;    // 4 slices x 512 x 2048 floats (16 MB)

  const size_t total_f = 1024 + (size_t)ROW_GROUPS * NCLS * DIM;

  k_init<<<2048, 256, 0, stream>>>(ws, total_f);
  k_counts<<<NUM / 256, 256, 0, stream>>>(labels, counts);
  k_main<<<dim3(N_TILES, ROW_GROUPS), 256, 0, stream>>>(x, labels, sums, accum);
  k_finalize<<<NCLS / 16, 256, 0, stream>>>(sums, counts, accum);
  k_write<<<1, 1, 0, stream>>>(accum, (float*)d_out);
}
